// LoRALinear_73976516706638
// MI455X (gfx1250) — matrix operand order repaired and checked by reading, not compiled, not run on hardware
//
#include <hip/hip_runtime.h>

// ---------------- types ----------------
typedef __attribute__((ext_vector_type(16))) __bf16 v16bf;
typedef __attribute__((ext_vector_type(8)))  __bf16 v8bf;
typedef __attribute__((ext_vector_type(8)))  float  v8f;
typedef __attribute__((ext_vector_type(4)))  float  v4f;
typedef __attribute__((ext_vector_type(4)))  int    v4i;
typedef __attribute__((ext_vector_type(4)))  unsigned int v4u;
typedef __attribute__((ext_vector_type(2)))  unsigned int v2u;

union FragBF { v16bf v; v8bf h[2]; };

#define M_TOT 8192      // B*S
#define N_TOT 4096      // O
#define K_TOT 4096      // I
#define BM 256          // block tile M (4 waves x 64 rows)
#define BN 128          // block tile N (2 waves x 64 cols)
#define BK 32
#define NBUF 4          // 4-deep LDS pipeline, prefetch distance 2
#define LDSW 40         // shorts per LDS tile row (32 data + 8 pad); 80B keeps 16B alignment

// pack two floats -> two bf16 in one dword (backend: v_cvt_pk_bf16_f32)
__device__ __forceinline__ unsigned int pack_bf16(float a, float b) {
  union { __bf16 h[2]; unsigned int u; } p;
  p.h[0] = (__bf16)a; p.h[1] = (__bf16)b;
  return p.u;
}

// ---- CDNA5 async global->LDS copy ----
#if __has_builtin(__builtin_amdgcn_global_load_async_to_lds_b128)
#define ASYNC_CP16(g, l)                                                   \
  __builtin_amdgcn_global_load_async_to_lds_b128((v4i*)(g), (v4i*)(l), 0, 0)
#else
#define ASYNC_CP16(g, l) do { *(v4u*)(l) = *(const v4u*)(g); } while (0)
#endif

#if __has_builtin(__builtin_amdgcn_s_wait_asynccnt)
#define WAIT_ASYNC(n) __builtin_amdgcn_s_wait_asynccnt(n)
#else
#define WAIT_ASYNC(n) asm volatile("s_wait_asynccnt %0" :: "i"(n) : "memory")
#endif

// ---------------------------------------------------------------------------
// Prologue: c[m] = sum_i x[m,i]*zero[i]*scale[i];  t[m,r] = sum_i x[m,i]*A[r,i]
// and emit bf16 copy of x (row-major [M,K]) for the WMMA GEMM.
// ---------------------------------------------------------------------------
__global__ __launch_bounds__(256)
void lora_prologue(const float* __restrict__ x, const float* __restrict__ scale,
                   const float* __restrict__ zero, const float* __restrict__ lA,
                   float* __restrict__ c_row, float* __restrict__ t_mat,
                   unsigned short* __restrict__ xb)
{
  const int wave = threadIdx.x >> 5;
  const int lane = threadIdx.x & 31;
  const int m = blockIdx.x * 8 + wave;
  const float* xr = x + (size_t)m * K_TOT;
  unsigned short* xo = xb + (size_t)m * K_TOT;

  float sc = 0.f;
  float st[8] = {0.f,0.f,0.f,0.f,0.f,0.f,0.f,0.f};
  for (int i = lane * 4; i < K_TOT; i += 32 * 4) {
    v4f xv = *(const v4f*)(xr + i);
    v4f zv = *(const v4f*)(zero + i);
    v4f sv = *(const v4f*)(scale + i);
    sc += xv[0]*zv[0]*sv[0] + xv[1]*zv[1]*sv[1] + xv[2]*zv[2]*sv[2] + xv[3]*zv[3]*sv[3];
#pragma unroll
    for (int r = 0; r < 8; ++r) {
      v4f av = *(const v4f*)(lA + (size_t)r * K_TOT + i);
      st[r] += xv[0]*av[0] + xv[1]*av[1] + xv[2]*av[2] + xv[3]*av[3];
    }
    v2u pk; pk[0] = pack_bf16(xv[0], xv[1]); pk[1] = pack_bf16(xv[2], xv[3]);
    *(v2u*)(xo + i) = pk;
  }
#pragma unroll
  for (int off = 16; off >= 1; off >>= 1) {
    sc += __shfl_xor(sc, off, 32);
#pragma unroll
    for (int r = 0; r < 8; ++r) st[r] += __shfl_xor(st[r], off, 32);
  }
  if (lane == 0) {
    c_row[m] = sc;
#pragma unroll
    for (int r = 0; r < 8; ++r) t_mat[m * 8 + r] = st[r];
  }
}

// ---------------------------------------------------------------------------
// One-time weight dequant: nibble -> bf16(q * scale[i]), row-major [O, I].
// 32 MiB result lives in L2 (192 MB) for the whole GEMM. One block per O-row.
// ---------------------------------------------------------------------------
__global__ __launch_bounds__(256)
void dequant_w(const int* __restrict__ wq, const float* __restrict__ scale,
               unsigned short* __restrict__ wb)
{
  const int row = blockIdx.x;                  // 0..4095
  const int j0  = threadIdx.x * 8;             // packed-int index within row
  const int* wp = wq + (size_t)row * (K_TOT / 2) + j0;
  v4i q0 = ((const v4i*)wp)[0], q1 = ((const v4i*)wp)[1];
  const float* sp = scale + 2 * j0;
  v4f s0 = ((const v4f*)sp)[0], s1 = ((const v4f*)sp)[1];
  v4f s2 = ((const v4f*)sp)[2], s3 = ((const v4f*)sp)[3];
  v4u o0, o1;
#pragma unroll
  for (int t = 0; t < 8; ++t) {
    const int q = (t < 4) ? q0[t] : q1[t - 4];
    const float slo = (t < 2 ? s0 : t < 4 ? s1 : t < 6 ? s2 : s3)[(2 * t) & 3];
    const float shi = (t < 2 ? s0 : t < 4 ? s1 : t < 6 ? s2 : s3)[(2 * t + 1) & 3];
    const unsigned int pk = pack_bf16((float)(q & 15) * slo, (float)((q >> 4) & 15) * shi);
    if (t < 4) o0[t] = pk; else o1[t - 4] = pk;
  }
  v4u* op = (v4u*)(wb + (size_t)row * K_TOT + 2 * j0);
  op[0] = o0; op[1] = o1;
}

// ---------------------------------------------------------------------------
// Main GEMM: 256x128 block tile, 64x64 per wave (16 WMMA / 16 frag-loads per
// K-step), async global->LDS 4-deep pipeline, one barrier per K-step.
// ---------------------------------------------------------------------------
__global__ __launch_bounds__(256)
void lora_w4_gemm(const unsigned short* __restrict__ xb,
                  const unsigned short* __restrict__ wb,
                  const float* __restrict__ bias, const float* __restrict__ lB,
                  const float* __restrict__ c_row, const float* __restrict__ t_mat,
                  float* __restrict__ out)
{
  __shared__ __attribute__((aligned(16))) unsigned short ldsA[NBUF][BM * LDSW]; // 4x20KB
  __shared__ __attribute__((aligned(16))) unsigned short ldsB[NBUF][BN * LDSW]; // 4x10KB

  const int tid  = threadIdx.x;
  const int bm   = blockIdx.y * BM;
  const int bn   = blockIdx.x * BN;
  const int wave = tid >> 5;
  const int lane = tid & 31;
  const int waveM = wave >> 1;        // 0..3 -> 64-row band
  const int waveN = wave & 1;         // 0..1 -> 64-col band
  const int half = lane >> 4;
  const int l16  = lane & 15;

  // staging: A = one 64B row per thread (4 chunks); B = 2 threads/row (2 chunks)
  const unsigned short* asrc_base = xb + (size_t)(bm + tid) * K_TOT;
  const unsigned short* bsrc_base = wb + (size_t)(bn + (tid >> 1)) * K_TOT + (tid & 1) * 16;
  unsigned short* adst = &ldsA[0][tid * LDSW];
  unsigned short* bdst = &ldsB[0][(tid >> 1) * LDSW + (tid & 1) * 16];

  auto load_tile = [&](int kt, int buf) {
    const unsigned short* as = asrc_base + kt * BK;
    const unsigned short* bs = bsrc_base + kt * BK;
    unsigned short* ad = adst + buf * (BM * LDSW);
    unsigned short* bd = bdst + buf * (BN * LDSW);
    ASYNC_CP16(as,      ad);
    ASYNC_CP16(as + 8,  ad + 8);
    ASYNC_CP16(as + 16, ad + 16);
    ASYNC_CP16(as + 24, ad + 24);
    ASYNC_CP16(bs,      bd);
    ASYNC_CP16(bs + 8,  bd + 8);
  };

  v8f acc[4][4] = {};
  const int nk = K_TOT / BK;          // 128

  auto step = [&](int kt, int cbuf, int pbuf) {
    const bool pf = (kt + 2 < nk);
    if (pf) load_tile(kt + 2, pbuf);  // async into buffer kt+2 (mod 4)

    FragBF af[4], bq[4];
#pragma unroll
    for (int mi = 0; mi < 4; ++mi) {  // A: 16B chunks at k = half*8 and 16+half*8
      const unsigned short* p = &ldsA[cbuf][(waveM * 64 + mi * 16 + l16) * LDSW + half * 8];
      af[mi].h[0] = *(const v8bf*)p;
      af[mi].h[1] = *(const v8bf*)(p + 16);
    }
#pragma unroll
    for (int ni = 0; ni < 4; ++ni) {  // B: col = l16, 16B chunks at k = half*8 and 16+half*8
      const unsigned short* p = &ldsB[cbuf][(waveN * 64 + ni * 16 + l16) * LDSW + half * 8];
      bq[ni].h[0] = *(const v8bf*)p;
      bq[ni].h[1] = *(const v8bf*)(p + 16);
    }
#pragma unroll
    for (int ni = 0; ni < 4; ++ni)
#pragma unroll
      for (int mi = 0; mi < 4; ++mi)
        acc[mi][ni] = __builtin_amdgcn_wmma_f32_16x16x32_bf16(
            false, af[mi].v, false, bq[ni].v, (short)0, acc[mi][ni], false, false);

    if (pf) WAIT_ASYNC(6);  // tile kt+1 landed; kt+2's 6 ops may stay in flight
    else    WAIT_ASYNC(0);
    __syncthreads();        // single barrier per K-step
  };

  // pipeline prologue: tiles 0 and 1 in flight
  load_tile(0, 0);
  load_tile(1, 1);
  WAIT_ASYNC(6);            // tile 0 landed
  __syncthreads();

  for (int kt = 0; kt < nk; kt += 4) {  // compile-time buffer ids
    step(kt,     0, 2);
    step(kt + 1, 1, 3);
    step(kt + 2, 2, 0);
    step(kt + 3, 3, 1);
  }

  // ---- epilogue: -c[m] + bias[n] + 2 * <t[m,:], lB[n,:]> ----
  const int om0 = bm + waveM * 64;
  const int on0 = bn + waveN * 64;
#pragma unroll
  for (int mi = 0; mi < 4; ++mi) {
    float cc[8]; v4f t0[8], t1[8];
#pragma unroll
    for (int j = 0; j < 8; ++j) {
      const int m = om0 + mi * 16 + half * 8 + j;   // C/D layout: VGPR j, lane-half
      cc[j] = c_row[m];
      const v4f* tp = (const v4f*)(t_mat + (size_t)m * 8);
      t0[j] = tp[0]; t1[j] = tp[1];
    }
#pragma unroll
    for (int ni = 0; ni < 4; ++ni) {
      const int n = on0 + ni * 16 + l16;
      const float bv = bias[n];
      const v4f* lp = (const v4f*)(lB + (size_t)n * 8);
      const v4f lb0 = lp[0], lb1 = lp[1];
#pragma unroll
      for (int j = 0; j < 8; ++j) {
        const int m = om0 + mi * 16 + half * 8 + j;
        float dot = t0[j][0]*lb0[0] + t0[j][1]*lb0[1] + t0[j][2]*lb0[2] + t0[j][3]*lb0[3]
                  + t1[j][0]*lb1[0] + t1[j][1]*lb1[1] + t1[j][2]*lb1[2] + t1[j][3]*lb1[3];
        out[(size_t)m * N_TOT + n] = acc[mi][ni][j] - cc[j] + bv + 2.0f * dot;
      }
    }
  }
}

// ---------------------------------------------------------------------------
extern "C" void kernel_launch(void* const* d_in, const int* in_sizes, int n_in,
                              void* d_out, int out_size, void* d_ws, size_t ws_size,
                              hipStream_t stream) {
  (void)in_sizes; (void)n_in; (void)out_size; (void)ws_size;
  const float* x      = (const float*)d_in[0];
  const int*   wq     = (const int*)d_in[1];
  const float* scale  = (const float*)d_in[2];
  const float* zero   = (const float*)d_in[3];
  const float* lora_A = (const float*)d_in[4];
  const float* lora_B = (const float*)d_in[5];
  const float* bias   = (const float*)d_in[6];
  float* out = (float*)d_out;

  // workspace layout: c_row (32 KiB) | t_mat (256 KiB) | x_bf16 (64 MiB) | w_bf16 (32 MiB)
  float* c_row = (float*)d_ws;
  float* t_mat = c_row + M_TOT;
  unsigned short* xbuf = (unsigned short*)(t_mat + (size_t)M_TOT * 8);
  unsigned short* wbuf = xbuf + (size_t)M_TOT * K_TOT;

  lora_prologue<<<M_TOT / 8, 256, 0, stream>>>(x, scale, zero, lora_A, c_row, t_mat, xbuf);
  dequant_w<<<N_TOT, 256, 0, stream>>>(wq, scale, wbuf);

  dim3 grid(N_TOT / BN, M_TOT / BM);      // (32, 32)
  lora_w4_gemm<<<grid, 256, 0, stream>>>(xbuf, wbuf, bias, lora_B, c_row, t_mat, out);
}
